// GATLayer_9509057593830
// MI455X (gfx1250) — matrix operand order repaired and checked by reading, not compiled, not run on hardware
//
#include <hip/hip_runtime.h>
#include <hip/hip_bf16.h>
#include <hip/hip_fp16.h>

// Problem sizes (fixed by reference)
#define BB   32
#define NN   512
#define INF  256     // input features
#define HH   8
#define DD   32
#define OUTF 256     // H*D
#define ROWS 16384   // B*N
#define NEG_SLOPE 0.2f

typedef __attribute__((ext_vector_type(16))) _Float16 v16h;
typedef __attribute__((ext_vector_type(8)))  _Float16 v8h;
typedef __attribute__((ext_vector_type(8)))  float    v8f;
typedef __attribute__((ext_vector_type(4)))  unsigned int v4u;
typedef __attribute__((ext_vector_type(8)))  int v8i;
typedef __attribute__((ext_vector_type(4)))  int v4i;

union AFrag { v16h v; v8h h[2]; };

struct __align__(8) H4 { _Float16 x, y, z, w; };

__device__ __forceinline__ float lrelu(float x) {
    return x > 0.0f ? x : NEG_SLOPE * x;
}

#if defined(__has_builtin)
# if __has_builtin(__builtin_amdgcn_tensor_load_to_lds)
#  define HAVE_TDM 1
# endif
#endif

// ---------------------------------------------------------------------------
// Kernel 1: Wx = x @ W^T   (M=16384, N=256, K=256), fp32 in, f16 out.
// Block tile 128x64, 256 threads = 8 waves, each wave owns a 16-row strip
// across all 4 N-tiles. Software-pipelined: next K-chunk is loaded into
// registers while the current chunk's 4 WMMAs execute.
// ---------------------------------------------------------------------------
#define XPAD 40   // halves per LDS row for X tile (32 + pad, 80B, 16B aligned)
#define WPAD 72   // halves per LDS row for W tile (64 + pad, 144B, 16B aligned)

__global__ __launch_bounds__(256)
void gemm_wx_kernel(const float* __restrict__ x, const float* __restrict__ W,
                    _Float16* __restrict__ Wxh) {
    __shared__ _Float16 Xs[128 * XPAD];       // [row][k] 128x32
    __shared__ _Float16 Ws[32 * WPAD];        // [k][n]   32x64 (transposed)

    const int tid  = threadIdx.x;
    const int lane = tid & 31;
    const int wave = tid >> 5;
    const int mBase = blockIdx.x * 128;
    const int nBase = blockIdx.y * 64;

    v8f acc0 = {}, acc1 = {}, acc2 = {}, acc3 = {};

    const int rowA = wave * 16 + (lane & 15);
    const int kb   = (lane >> 4) * 8;

    // staging coordinates (fixed per thread)
    int xr[4], xc[4];
    #pragma unroll
    for (int i = 0; i < 4; ++i) { int q = tid + 256 * i; xr[i] = q >> 3; xc[i] = (q & 7) * 4; }
    int wn[2], wc[2];
    #pragma unroll
    for (int i = 0; i < 2; ++i) { int q = tid + 256 * i; wn[i] = q >> 3; wc[i] = (q & 7) * 4; }

    // ---- preload chunk 0 into registers ----
    float4 fx[4], fw[2];
    #pragma unroll
    for (int i = 0; i < 4; ++i)
        fx[i] = *(const float4*)(x + (size_t)(mBase + xr[i]) * INF + xc[i]);
    #pragma unroll
    for (int i = 0; i < 2; ++i)
        fw[i] = *(const float4*)(W + (size_t)(nBase + wn[i]) * INF + wc[i]);

    for (int kc = 0; kc < 8; ++kc) {
        // ---- convert + store the staged registers to LDS ----
        #pragma unroll
        for (int i = 0; i < 4; ++i) {
            H4 hv = { (_Float16)fx[i].x, (_Float16)fx[i].y, (_Float16)fx[i].z, (_Float16)fx[i].w };
            *(H4*)&Xs[xr[i] * XPAD + xc[i]] = hv;
        }
        #pragma unroll
        for (int i = 0; i < 2; ++i) {
            Ws[(wc[i] + 0) * WPAD + wn[i]] = (_Float16)fw[i].x;
            Ws[(wc[i] + 1) * WPAD + wn[i]] = (_Float16)fw[i].y;
            Ws[(wc[i] + 2) * WPAD + wn[i]] = (_Float16)fw[i].z;
            Ws[(wc[i] + 3) * WPAD + wn[i]] = (_Float16)fw[i].w;
        }
        __syncthreads();

        // ---- issue next chunk's global loads (hidden under the WMMAs) ----
        if (kc + 1 < 8) {
            const int kOff = (kc + 1) * 32;
            #pragma unroll
            for (int i = 0; i < 4; ++i) {
                const float* p = x + (size_t)(mBase + xr[i]) * INF + kOff + xc[i];
                fx[i] = *(const float4*)p;
                if (kc + 2 < 8)   // keep a prefetch stream one chunk ahead
                    __builtin_prefetch(p + 32, 0, 1);
            }
            #pragma unroll
            for (int i = 0; i < 2; ++i)
                fw[i] = *(const float4*)(W + (size_t)(nBase + wn[i]) * INF + kOff + wc[i]);
        }

        // ---- A fragment: lane l holds row (l&15), K = {kb..kb+7, 16+kb..16+kb+7}
        AFrag a;
        a.h[0] = *(const v8h*)&Xs[rowA * XPAD + kb];
        a.h[1] = *(const v8h*)&Xs[rowA * XPAD + 16 + kb];

        // ---- B fragments: lane l holds column N=(l&15), K = {kb..kb+7, 16+kb..16+kb+7}
        #pragma unroll
        for (int t = 0; t < 4; ++t) {
            AFrag bfr;
            #pragma unroll
            for (int s = 0; s < 16; ++s) bfr.v[s] = Ws[((s < 8) ? (kb + s) : (16 + kb + s - 8)) * WPAD + t * 16 + (lane & 15)];
            if (t == 0) acc0 = __builtin_amdgcn_wmma_f32_16x16x32_f16(false, a.v, false, bfr.v, (short)0, acc0, false, false);
            if (t == 1) acc1 = __builtin_amdgcn_wmma_f32_16x16x32_f16(false, a.v, false, bfr.v, (short)0, acc1, false, false);
            if (t == 2) acc2 = __builtin_amdgcn_wmma_f32_16x16x32_f16(false, a.v, false, bfr.v, (short)0, acc2, false, false);
            if (t == 3) acc3 = __builtin_amdgcn_wmma_f32_16x16x32_f16(false, a.v, false, bfr.v, (short)0, acc3, false, false);
        }
        __syncthreads();
    }

    // ---- epilogue: C/D layout lane l, vgpr v -> row v+8*(l>>4), col l&15 ----
    const int colL = lane & 15;
    const int rHi  = 8 * (lane >> 4);
    #pragma unroll
    for (int v = 0; v < 8; ++v) {
        int grow = mBase + wave * 16 + v + rHi;
        size_t base = (size_t)grow * OUTF + nBase + colL;
        Wxh[base +  0] = (_Float16)acc0[v];
        Wxh[base + 16] = (_Float16)acc1[v];
        Wxh[base + 32] = (_Float16)acc2[v];
        Wxh[base + 48] = (_Float16)acc3[v];
    }
}

// ---------------------------------------------------------------------------
// Kernel 2: per-(row,h) score dots: s_src = Wx[row,h,:]·a_src, s_tgt likewise
// ---------------------------------------------------------------------------
__global__ __launch_bounds__(256)
void scores_kernel(const _Float16* __restrict__ Wxh, const float* __restrict__ attn,
                   float* __restrict__ s_src, float* __restrict__ s_tgt) {
    int g   = blockIdx.x * 256 + threadIdx.x;   // 0 .. ROWS*H-1
    int row = g >> 3;
    int h   = g & 7;
    const _Float16* wp = Wxh + (size_t)row * OUTF + h * DD;
    const float* asrc = attn + h * (2 * DD);
    const float* atgt = asrc + DD;
    float ss = 0.0f, st = 0.0f;
    #pragma unroll
    for (int c = 0; c < 4; ++c) {
        v8h w = *(const v8h*)(wp + c * 8);
        #pragma unroll
        for (int u = 0; u < 8; ++u) {
            float wv = (float)w[u];
            ss += wv * asrc[c * 8 + u];
            st += wv * atgt[c * 8 + u];
        }
    }
    s_src[(size_t)row * HH + h] = ss;
    s_tgt[(size_t)row * HH + h] = st;
}

// ---------------------------------------------------------------------------
// Kernel 3: per-(b,h) attention. Wx head slice is staged to LDS via the
// Tensor Data Mover (2D tile, 512 rows x 64B, TDM padding recreates the 80B
// LDS row stride); P tiles are built per-lane in the WMMA A-layout and fed
// to v_wmma_f32_16x16x32_f16; denominator via shfl_xor(16); /den + elu.
// ---------------------------------------------------------------------------
#define WXPAD 40   // halves per LDS row (32 + 8 pad halves = 80B)

__global__ __launch_bounds__(256)
void attn_out_kernel(const _Float16* __restrict__ Wxh,
                     const float* __restrict__ s_src_g,
                     const float* __restrict__ s_tgt_g,
                     float* __restrict__ out) {
    __shared__ _Float16 Wxs[NN * WXPAD];   // [j][d] 512x32 head slice
    __shared__ float ss[NN];
    __shared__ float st[NN];
    __shared__ float red[256];

    const int tid  = threadIdx.x;
    const int lane = tid & 31;
    const int wave = tid >> 5;
    const int b = blockIdx.x >> 3;
    const int h = blockIdx.x & 7;

#if defined(HAVE_TDM)
    // ---- TDM: DMA the 512x32 f16 head slice into LDS with padded rows ----
    if (wave == 0) {
        unsigned lds_off = (unsigned)(uintptr_t)&Wxs[0];   // LDS aperture: low 32 bits = offset
        unsigned long long ga =
            (unsigned long long)(uintptr_t)(Wxh + (size_t)(b * NN) * OUTF + h * DD);
        // D# group 0: count=1 | lds_addr | 57-bit global addr | type=2
        v4u g0 = { 1u,
                   lds_off,
                   (unsigned)(ga & 0xFFFFFFFFu),
                   (unsigned)((ga >> 32) & 0x1FFFFFFu) | (2u << 30) };
        // D# group 1: data_size=2B, pad_enable, pad_interval=16 DWORDs (code 3),
        // pad_amount=4 DWORDs (code 3); tensor 32x512, tile 32x512, stride0=256.
        v8i g1 = { (int)((1u << 16) | (1u << 20) | (3u << 22) | (3u << 25)),
                   (int)(32u << 16),            // tensor_dim0 = 32 (low16)
                   (int)(512u << 16),           // tensor_dim0 hi | tensor_dim1 = 512 (low16)
                   (int)(32u << 16),            // tensor_dim1 hi | tile_dim0 = 32
                   (int)512,                    // tile_dim1 = 512, tile_dim2 = 0
                   (int)256,                    // tensor_dim0_stride = 256 (low32)
                   0, 0 };
        v4i gz = { 0, 0, 0, 0 };
# if __clang_major__ >= 23
        v8i gz8 = { 0, 0, 0, 0, 0, 0, 0, 0 };
        __builtin_amdgcn_tensor_load_to_lds(g0, g1, gz, gz, gz8, 0);
# else
        __builtin_amdgcn_tensor_load_to_lds(g0, g1, gz, gz, 0);
# endif
        __builtin_amdgcn_s_wait_tensorcnt(0);
    }
#else
    // ---- fallback: cooperative uint4 staging ----
    #pragma unroll
    for (int i = 0; i < 8; ++i) {
        int q  = tid + 256 * i;          // 0..2047 uint4 slots
        int j  = q >> 2;
        int c4 = q & 3;
        const uint4* src = (const uint4*)(Wxh + (size_t)(b * NN + j) * OUTF + h * DD) + c4;
        *(uint4*)&Wxs[j * WXPAD + c4 * 8] = *src;
    }
#endif
    // ---- stage score vectors ----
    ss[tid]       = s_src_g[(size_t)(b * NN + tid) * HH + h];
    ss[tid + 256] = s_src_g[(size_t)(b * NN + tid + 256) * HH + h];
    st[tid]       = s_tgt_g[(size_t)(b * NN + tid) * HH + h];
    st[tid + 256] = s_tgt_g[(size_t)(b * NN + tid + 256) * HH + h];
    __syncthreads();

    // ---- block-wide max over s_tgt (monotone lrelu => row max from this) ----
    red[tid] = fmaxf(st[tid], st[tid + 256]);
    __syncthreads();
    for (int s = 128; s > 0; s >>= 1) {
        if (tid < s) red[tid] = fmaxf(red[tid], red[tid + s]);
        __syncthreads();
    }
    const float mtgt = red[0];

    const int iloc = lane & 15;
    const int kb   = (lane >> 4) * 8;
    const int rHi  = 8 * (lane >> 4);

    // 32 i-tiles / 8 waves = 4 tiles per wave
    #pragma unroll 1
    for (int tt = 0; tt < 4; ++tt) {
        const int it = wave * 4 + tt;
        const float ssrc_i  = ss[it * 16 + iloc];
        const float rowmax  = lrelu(ssrc_i + mtgt);

        v8f acc0 = {}, acc1 = {};
        float densum = 0.0f;

        for (int c = 0; c < 16; ++c) {
            // build P fragment in A layout: elems 0..7 -> K=kb..kb+7,
            // elems 8..15 -> K=16+kb..16+kb+7 (K = j within chunk)
            AFrag a;
            #pragma unroll
            for (int u = 0; u < 8; ++u) {
                int j0 = c * 32 + kb + u;
                int j1 = c * 32 + 16 + kb + u;
                float p0 = __expf(lrelu(ssrc_i + st[j0]) - rowmax);
                float p1 = __expf(lrelu(ssrc_i + st[j1]) - rowmax);
                densum += p0 + p1;
                a.v[u]     = (_Float16)p0;
                a.v[u + 8] = (_Float16)p1;
            }
            // B fragments: lane l -> d col (l&15), K = j in same order as A
            AFrag b0, b1;
            #pragma unroll
            for (int s = 0; s < 16; ++s) {
                const _Float16* brow = &Wxs[(c * 32 + ((s < 8) ? (kb + s) : (16 + kb + s - 8))) * WXPAD + iloc];
                b0.v[s] = brow[0];
                b1.v[s] = brow[16];
            }
            acc0 = __builtin_amdgcn_wmma_f32_16x16x32_f16(false, a.v, false, b0.v, (short)0, acc0, false, false);
            acc1 = __builtin_amdgcn_wmma_f32_16x16x32_f16(false, a.v, false, b1.v, (short)0, acc1, false, false);
        }

        // denominator: lane l and l+16 jointly cover all j for row l&15
        float den_i = densum + __shfl_xor(densum, 16, 32);

        // epilogue: divide by den of row M, apply elu, store
        #pragma unroll
        for (int v = 0; v < 8; ++v) {
            int   M  = v + rHi;
            float dv = __shfl(den_i, M, 32);   // lane M holds den of row M
            int grow = it * 16 + M;
            float o0 = acc0[v] / dv;
            float o1 = acc1[v] / dv;
            o0 = o0 > 0.0f ? o0 : (__expf(o0) - 1.0f);
            o1 = o1 > 0.0f ? o1 : (__expf(o1) - 1.0f);
            size_t base = ((size_t)b * NN + grow) * OUTF + h * DD + iloc;
            out[base]      = o0;
            out[base + 16] = o1;
        }
    }
}

// ---------------------------------------------------------------------------
extern "C" void kernel_launch(void* const* d_in, const int* in_sizes, int n_in,
                              void* d_out, int out_size, void* d_ws, size_t ws_size,
                              hipStream_t stream) {
    const float* x    = (const float*)d_in[0];   // (32,512,256)
    const float* W    = (const float*)d_in[1];   // (256,256)
    const float* attn = (const float*)d_in[2];   // (1,8,64)
    float* out = (float*)d_out;                  // (32,512,256)

    char* ws = (char*)d_ws;
    _Float16* Wxh = (_Float16*)ws;                               // 16384*256 f16 = 8 MB
    float* s_src  = (float*)(ws + (size_t)ROWS * OUTF * 2);      // 16384*8 f32
    float* s_tgt  = s_src + (size_t)ROWS * HH;

    // 1) Wx = x @ W^T (f16 out)
    gemm_wx_kernel<<<dim3(ROWS / 128, OUTF / 64), 256, 0, stream>>>(x, W, Wxh);
    // 2) attention score dots
    scores_kernel<<<dim3((ROWS * HH) / 256), 256, 0, stream>>>(Wxh, attn, s_src, s_tgt);
    // 3) fused softmax-attention GEMM + elu (TDM-staged LDS)
    attn_out_kernel<<<dim3(BB * HH), 256, 0, stream>>>(Wxh, s_src, s_tgt, out);
}
